// CRF_83880711291182
// MI455X (gfx1250) — matrix operand order repaired and checked by reading, not compiled, not run on hardware
//
#include <hip/hip_runtime.h>
#include <hip/hip_bf16.h>

typedef __attribute__((ext_vector_type(16))) _Float16 v16h;
typedef __attribute__((ext_vector_type(8)))  _Float16 v8h;
typedef __attribute__((ext_vector_type(8)))  float    v8f;

#define B_SZ 1024
#define S_SZ 512
#define T_SZ 48

#define LOG2E 1.44269504088896340736f
#define LN2   0.69314718055994530942f

// Fast transcendentals: accumulator is provably in [exp(-0.1), ~53], and exp args
// are <= 0, so raw v_exp_f32/v_log_f32 (no denorm fixup) are safe.
__device__ __forceinline__ float fast_exp(float x) {
    return __builtin_amdgcn_exp2f(x * LOG2E);
}
__device__ __forceinline__ float fast_log(float x) {
    return __builtin_amdgcn_logf(x) * LN2;
}

// ---------------------------------------------------------------------------
// Forward algorithm (log-partition) — 16 chains per wave, WMMA f32_16x16x32_f16.
//
// Per step:  alpha'[c][j] = log( sum_i exp(alpha[c][i]-m_c) * E[i][j] ) + m_c + em[c][t][j]
// with E = exp(transitions) held as pre-swizzled WMMA B-operands in VGPRs.
//
// C/D layout (32-bit, 16x16): lane L holds column n=L%16, rows m=(L/16)*8+v in VGPR v.
// A layout  (16-bit, 16x32):  lane L holds row m=L%16, K = (L/16)*8+{0..7} in elems 0..7
//                             and K = 16+(L/16)*8+{0..7} in elems 8..15.
// B layout  (16-bit, 32x16):  lane L holds column n=L%16, K=(L/16)*8+e in elems 0..7, K=16+(L/16)*8+(e-8) in elems 8..15.
// ---------------------------------------------------------------------------
__global__ __launch_bounds__(32)
void crf_forward_logZ(const float* __restrict__ em,
                      const unsigned char* __restrict__ mask,
                      const float* __restrict__ start_t,
                      const float* __restrict__ end_t,
                      const float* __restrict__ trans,
                      float* __restrict__ logZ_out)
{
    __shared__ __align__(16) _Float16 pbuf[16][72];   // 16 chains x 64 states (+8 pad)

    const int lane  = threadIdx.x;   // 0..31
    const int nl    = lane & 15;     // column / state-within-tile index
    const int h     = lane >> 4;     // half-wave select
    const int cbase = blockIdx.x * 16;

    // Zero LDS once: states 48..63 act as K-padding for the WMMA and must stay 0.
    {
        _Float16* pb = &pbuf[0][0];
        for (int i = lane; i < 16 * 72; i += 32) pb[i] = (_Float16)0.0f;
    }

    // Precompute E = exp(transitions) directly in WMMA B-operand layout.
    v16h Eop[3][2];
    #pragma unroll
    for (int t3 = 0; t3 < 3; ++t3) {
        const int n = t3 * 16 + nl;                 // n < 48 always
        #pragma unroll
        for (int sl = 0; sl < 2; ++sl) {
            v16h e;
            #pragma unroll
            for (int kk = 0; kk < 16; ++kk) {
                const int K = sl * 32 + (kk < 8 ? h * 8 + kk : 16 + h * 8 + (kk - 8));
                const float v = (K < T_SZ) ? fast_exp(trans[K * T_SZ + n]) : 0.0f;
                e[kk] = (_Float16)v;
            }
            Eop[t3][sl] = e;
        }
    }

    // alpha in C-layout: alpha[t3][v] = chain (h*8+v), state (t3*16+nl)
    float alpha[3][8];
    #pragma unroll
    for (int t3 = 0; t3 < 3; ++t3) {
        const int s = t3 * 16 + nl;
        const float st = start_t[s];
        #pragma unroll
        for (int v = 0; v < 8; ++v) {
            const int c = cbase + h * 8 + v;
            alpha[t3][v] = st + em[(size_t)c * S_SZ * T_SZ + s];
        }
    }

    // Induction bases: per-v / per-tile deltas become 24-bit instruction immediates.
    const float* emb = em + (size_t)(cbase + h * 8) * S_SZ * T_SZ + nl;
    const unsigned char* mkb = mask + (size_t)(cbase + h * 8) * S_SZ;

    for (int t = 1; t < S_SZ; ++t) {
        // ---- stream emissions + mask for this step ----
        const float* ept = emb + (size_t)t * T_SZ;
        float emv[3][8];
        float mskv[8];
        #pragma unroll
        for (int v = 0; v < 8; ++v) {
            emv[0][v] = ept[v * (S_SZ * T_SZ)];
            emv[1][v] = ept[v * (S_SZ * T_SZ) + 16];
            emv[2][v] = ept[v * (S_SZ * T_SZ) + 32];
            mskv[v] = (float)mkb[v * S_SZ + t];     // bool byte -> 0.0f / 1.0f
        }

        // ---- per-chain max (reduce across the 16 lanes of each half-wave) ----
        float pm[8];
        #pragma unroll
        for (int v = 0; v < 8; ++v) {
            float x = fmaxf(fmaxf(alpha[0][v], alpha[1][v]), alpha[2][v]);
            x = fmaxf(x, __shfl_xor(x, 1, 32));
            x = fmaxf(x, __shfl_xor(x, 2, 32));
            x = fmaxf(x, __shfl_xor(x, 4, 32));
            x = fmaxf(x, __shfl_xor(x, 8, 32));
            pm[v] = x;
        }

        // ---- p = exp(alpha - m) -> LDS (C-layout -> chain-major f16) ----
        #pragma unroll
        for (int t3 = 0; t3 < 3; ++t3)
            #pragma unroll
            for (int v = 0; v < 8; ++v)
                pbuf[h * 8 + v][t3 * 16 + nl] = (_Float16)fast_exp(alpha[t3][v] - pm[v]);

        // ---- reload as WMMA A-operands (wave-synchronous: DS ops in-order in-wave) ----
        const int m = nl;
        const v8h a0lo = *(const v8h*)&pbuf[m][h * 8];        // K =      h*8 .. +7
        const v8h a0hi = *(const v8h*)&pbuf[m][16 + h * 8];   // K = 16 + h*8 .. +7
        const v8h a1lo = *(const v8h*)&pbuf[m][32 + h * 8];   // K = 32 + h*8 .. +7
        const v8h a1hi = *(const v8h*)&pbuf[m][48 + h * 8];   // K = 48.. (zero pad)
        const v16h A0 = __builtin_shufflevector(a0lo, a0hi, 0,1,2,3,4,5,6,7,8,9,10,11,12,13,14,15);
        const v16h A1 = __builtin_shufflevector(a1lo, a1hi, 0,1,2,3,4,5,6,7,8,9,10,11,12,13,14,15);

        // ---- (16x64) @ (64x48): issue all 6 WMMAs before any VALU consumption ----
        v8f acc0 = {}, acc1 = {}, acc2 = {};
        acc0 = __builtin_amdgcn_wmma_f32_16x16x32_f16(false, A0, false, Eop[0][0], (short)0, acc0, false, false);
        acc1 = __builtin_amdgcn_wmma_f32_16x16x32_f16(false, A0, false, Eop[1][0], (short)0, acc1, false, false);
        acc2 = __builtin_amdgcn_wmma_f32_16x16x32_f16(false, A0, false, Eop[2][0], (short)0, acc2, false, false);
        acc0 = __builtin_amdgcn_wmma_f32_16x16x32_f16(false, A1, false, Eop[0][1], (short)0, acc0, false, false);
        acc1 = __builtin_amdgcn_wmma_f32_16x16x32_f16(false, A1, false, Eop[1][1], (short)0, acc1, false, false);
        acc2 = __builtin_amdgcn_wmma_f32_16x16x32_f16(false, A1, false, Eop[2][1], (short)0, acc2, false, false);

        // ---- pe = pm + em precomputed (independent of WMMA results, overlaps them) ----
        float pe0[8], pe1[8], pe2[8];
        #pragma unroll
        for (int v = 0; v < 8; ++v) {
            pe0[v] = pm[v] + emv[0][v];
            pe1[v] = pm[v] + emv[1][v];
            pe2[v] = pm[v] + emv[2][v];
        }

        // ---- alpha' = mask-blend(fma(log2(acc), ln2, pe)), branchless ----
        #pragma unroll
        for (int v = 0; v < 8; ++v) {
            const float a0 = fmaf(__builtin_amdgcn_logf(acc0[v]), LN2, pe0[v]);
            const float a1 = fmaf(__builtin_amdgcn_logf(acc1[v]), LN2, pe1[v]);
            const float a2 = fmaf(__builtin_amdgcn_logf(acc2[v]), LN2, pe2[v]);
            alpha[0][v] = fmaf(mskv[v], a0 - alpha[0][v], alpha[0][v]);
            alpha[1][v] = fmaf(mskv[v], a1 - alpha[1][v], alpha[1][v]);
            alpha[2][v] = fmaf(mskv[v], a2 - alpha[2][v], alpha[2][v]);
        }
    }

    // ---- logZ = logsumexp_j(alpha[j] + end[j]) per chain ----
    #pragma unroll
    for (int t3 = 0; t3 < 3; ++t3) {
        const float e = end_t[t3 * 16 + nl];
        #pragma unroll
        for (int v = 0; v < 8; ++v) alpha[t3][v] += e;
    }
    #pragma unroll
    for (int v = 0; v < 8; ++v) {
        float x = fmaxf(fmaxf(alpha[0][v], alpha[1][v]), alpha[2][v]);
        x = fmaxf(x, __shfl_xor(x, 1, 32));
        x = fmaxf(x, __shfl_xor(x, 2, 32));
        x = fmaxf(x, __shfl_xor(x, 4, 32));
        x = fmaxf(x, __shfl_xor(x, 8, 32));
        float s = fast_exp(alpha[0][v] - x) + fast_exp(alpha[1][v] - x) + fast_exp(alpha[2][v] - x);
        s += __shfl_xor(s, 1, 32);
        s += __shfl_xor(s, 2, 32);
        s += __shfl_xor(s, 4, 32);
        s += __shfl_xor(s, 8, 32);
        if (nl == 0) logZ_out[cbase + h * 8 + v] = x + fast_log(s);
    }
}

// ---------------------------------------------------------------------------
// Gold (numerator) score: one thread per chain.
// ---------------------------------------------------------------------------
__global__ __launch_bounds__(256)
void crf_gold(const float* __restrict__ em,
              const long long* __restrict__ tags,
              const unsigned char* __restrict__ mask,
              const float* __restrict__ start_t,
              const float* __restrict__ end_t,
              const float* __restrict__ trans,
              float* __restrict__ score_out)
{
    const int c = blockIdx.x * 256 + threadIdx.x;
    if (c >= B_SZ) return;
    const long long* tg = tags + (size_t)c * S_SZ;
    const unsigned char* mk = mask + (size_t)c * S_SZ;
    const float* eb = em + (size_t)c * S_SZ * T_SZ;

    int prev = (int)tg[0];
    float score = start_t[prev] + eb[prev];
    int seq_end = 0;  // = sum(mask) - 1 given mask[0] == true
    for (int t = 1; t < S_SZ; ++t) {
        const int cur = (int)tg[t];
        const float mf = (float)mk[t];
        score += mf * (trans[prev * T_SZ + cur] + eb[(size_t)t * T_SZ + cur]);
        seq_end += mk[t] ? 1 : 0;
        prev = cur;
    }
    score += end_t[(int)tg[seq_end]];
    score_out[c] = score;
}

// ---------------------------------------------------------------------------
// Deterministic fixed-order final reduction: out = sum_c (score[c] - logZ[c]).
// ---------------------------------------------------------------------------
__global__ __launch_bounds__(256)
void crf_reduce(const float* __restrict__ score,
                const float* __restrict__ logZ,
                float* __restrict__ out)
{
    __shared__ float sm[256];
    const int tid = threadIdx.x;
    float s = 0.0f;
    #pragma unroll
    for (int i = 0; i < 4; ++i) {
        const int c = tid + i * 256;
        s += score[c] - logZ[c];
    }
    sm[tid] = s;
    __syncthreads();
    for (int w = 128; w > 0; w >>= 1) {
        if (tid < w) sm[tid] += sm[tid + w];
        __syncthreads();
    }
    if (tid == 0) out[0] = sm[0];
}

extern "C" void kernel_launch(void* const* d_in, const int* in_sizes, int n_in,
                              void* d_out, int out_size, void* d_ws, size_t ws_size,
                              hipStream_t stream) {
    const float*         em      = (const float*)d_in[0];
    const long long*     tags    = (const long long*)d_in[1];
    const unsigned char* mask    = (const unsigned char*)d_in[2];
    const float*         start_t = (const float*)d_in[3];
    const float*         end_t   = (const float*)d_in[4];
    const float*         trans   = (const float*)d_in[5];

    float* score_ws = (float*)d_ws;          // [B_SZ]
    float* logZ_ws  = score_ws + B_SZ;       // [B_SZ]
    float* out      = (float*)d_out;

    crf_forward_logZ<<<B_SZ / 16, 32, 0, stream>>>(em, mask, start_t, end_t, trans, logZ_ws);
    crf_gold<<<B_SZ / 256, 256, 0, stream>>>(em, tags, mask, start_t, end_t, trans, score_ws);
    crf_reduce<<<1, 256, 0, stream>>>(score_ws, logZ_ws, out);
}
